// CrossAttention_73529840107646
// MI455X (gfx1250) — matrix operand order repaired and checked by reading, not compiled, not run on hardware
//
#include <hip/hip_runtime.h>
#include <hip/hip_bf16.h>

#define DIM     1024
#define NHEADS  16
#define HDIM    64
#define BATCH   4
#define NQ      2048
#define NK      2048
#define SCALE   0.125f      /* 64^-0.5 */
#define NEG_INF -1e9f

typedef __attribute__((ext_vector_type(16))) __bf16        v16bf;
typedef __attribute__((ext_vector_type(2)))  __bf16        v2bf;
typedef __attribute__((ext_vector_type(8)))  float         v8f;
typedef __attribute__((ext_vector_type(4)))  float         v4f;
typedef __attribute__((ext_vector_type(4)))  unsigned int  v4u;
typedef __attribute__((ext_vector_type(4)))  int           v4i;

union Frag16 { v16bf v; unsigned short us[16]; v4u q[2]; };

__device__ __forceinline__ unsigned short bfbits(float f) {
  __bf16 h = (__bf16)f;
  return __builtin_bit_cast(unsigned short, h);
}

__device__ __forceinline__ unsigned int pack2(float a, float b) {
  union { v2bf v; unsigned int u; } c;
  c.v[0] = (__bf16)a;
  c.v[1] = (__bf16)b;
  return c.u;
}

// ---------------------------------------------------------------------------
// Streaming fp32 -> bf16 conversion (one pass; hoists all cvt VALU work out
// of the GEMM inner loops).  n must be a multiple of 8.
// ---------------------------------------------------------------------------
__global__ __launch_bounds__(256) void cvt_bf16_kernel(
    const float* __restrict__ X, unsigned short* __restrict__ Y, int n8)
{
  int i = blockIdx.x * blockDim.x + threadIdx.x;
  if (i >= n8) return;
  const float* x = X + (size_t)i * 8;
  v4f a = *(const v4f*)(x);
  v4f b = *(const v4f*)(x + 4);
  v4u o;
  o[0] = pack2(a[0], a[1]);
  o[1] = pack2(a[2], a[3]);
  o[2] = pack2(b[0], b[1]);
  o[3] = pack2(b[2], b[3]);
  *(v4u*)(Y + (size_t)i * 8) = o;
}

// ---------------------------------------------------------------------------
// Unified bf16 WMMA GEMM: Y = A[8192x1024] @ W[1024x1024]^T
// One wave computes a 32(M) x 64(N) tile: 2 A-frags x 4 B-frags -> 8 WMMA
// per 32-deep k-step.  Inner loop is pure b128 loads + v_wmma.
// mode 0: bf16 out, [B,H,N,64] layout (Q/K heads)
// mode 1: bf16 out, [B,H,64,Nk] layout (V transposed)
// mode 2: fp32 out, row-major [8192,1024], += bias
// ---------------------------------------------------------------------------
__global__ __launch_bounds__(256) void gemm_bf16_kernel(
    const unsigned short* __restrict__ A, const unsigned short* __restrict__ W,
    unsigned short* __restrict__ Ybf, float* __restrict__ Yf,
    const float* __restrict__ bias, int mode)
{
  const int lane = threadIdx.x & 31;
  const int wave = blockIdx.x * (blockDim.x >> 5) + (threadIdx.x >> 5);
  const int m0   = (wave >> 4) * 32;   // 256 row tiles
  const int n0   = (wave & 15) * 64;   // 16 col tiles
  const int half = lane >> 4;
  const int l15  = lane & 15;
  const int akb  = half ? 8  : 0;      // A-frag K base (ISA 16-bit A layout)
  const int bkb  = half ? 8  : 0;      // B-frag K base (ISA 16-bit layout, same as A)

  v8f acc[2][4];
  #pragma unroll
  for (int i = 0; i < 2; ++i)
    #pragma unroll
    for (int j = 0; j < 4; ++j) acc[i][j] = {};

  for (int k0 = 0; k0 < DIM; k0 += 32) {
    Frag16 a[2], b[4];
    #pragma unroll
    for (int ci = 0; ci < 2; ++ci) {
      const unsigned short* ar = A + (size_t)(m0 + 16 * ci + l15) * DIM + k0 + akb;
      a[ci].q[0] = *(const v4u*)(ar);
      a[ci].q[1] = *(const v4u*)(ar + 16);
    }
    #pragma unroll
    for (int cj = 0; cj < 4; ++cj) {
      const unsigned short* wr = W + (size_t)(n0 + 16 * cj + l15) * DIM + k0 + bkb;
      b[cj].q[0] = *(const v4u*)(wr);
      b[cj].q[1] = *(const v4u*)(wr + 16);
    }
    #pragma unroll
    for (int ci = 0; ci < 2; ++ci)
      #pragma unroll
      for (int cj = 0; cj < 4; ++cj)
        acc[ci][cj] = __builtin_amdgcn_wmma_f32_16x16x32_bf16(
            false, a[ci].v, false, b[cj].v, (short)0, acc[ci][cj], false, false);
  }

  #pragma unroll
  for (int ci = 0; ci < 2; ++ci)
    #pragma unroll
    for (int cj = 0; cj < 4; ++cj) {
      const int col = n0 + 16 * cj + l15;
      if (mode == 2) {
        const float bb = bias[col];
        #pragma unroll
        for (int r = 0; r < 8; ++r) {
          int row = m0 + 16 * ci + r + 8 * half;
          Yf[(size_t)row * DIM + col] = acc[ci][cj][r] + bb;
        }
      } else {
        const int h = col >> 6, dh = col & 63;
        #pragma unroll
        for (int r = 0; r < 8; ++r) {
          int row = m0 + 16 * ci + r + 8 * half;   // b*N + n
          int bb = row >> 11, n = row & 2047;
          size_t idx = (mode == 0)
              ? (((size_t)(bb * NHEADS + h) * NQ)   + n)  * HDIM + dh
              : (((size_t)(bb * NHEADS + h) * HDIM) + dh) * NK   + n;
          Ybf[idx] = bfbits(acc[ci][cj][r]);
        }
      }
    }
}

// ---------------------------------------------------------------------------
// Flash attention, one wave per 16-query tile, online softmax.
// Computes S^T = K*Q^T so softmax stats are per-lane scalars (query = lane&15)
// and O^T = V^T * P^T so the accumulator rescale is a uniform per-lane mul.
// ---------------------------------------------------------------------------
__global__ __launch_bounds__(256) void attn_kernel(
    const unsigned short* __restrict__ Qh,  // [B,H,NQ,64] bf16
    const unsigned short* __restrict__ Kh,  // [B,H,NK,64] bf16
    const unsigned short* __restrict__ Vt,  // [B,H,64,NK] bf16
    const int* __restrict__ mask,           // [B,NK]
    unsigned short* __restrict__ O)         // [B,NQ,DIM] bf16
{
  const int lane = threadIdx.x & 31;
  const int wave = blockIdx.x * (blockDim.x >> 5) + (threadIdx.x >> 5);
  const int qt   = wave & (NQ / 16 - 1);
  const int bh   = wave >> 7;
  const int b    = bh >> 4;
  const int h    = bh & 15;
  const int q0   = qt * 16;
  const int half = lane >> 4;
  const int l15  = lane & 15;
  const int akb  = half ? 8 : 0;

  const unsigned short* Qbase = Qh + (size_t)(b * NHEADS + h) * NQ * HDIM;
  const unsigned short* Kbase = Kh + (size_t)(b * NHEADS + h) * NK * HDIM;
  const unsigned short* Vbase = Vt + (size_t)(b * NHEADS + h) * HDIM * NK;
  const int* mrow = mask + b * NK;

  // Q as B-fragments over head-dim (ISA 16-bit layout, same K order as A)
  Frag16 qb[2];
  {
    const unsigned short* qr = Qbase + (size_t)(q0 + l15) * HDIM + (half ? 8 : 0);
    qb[0].q[0] = *(const v4u*)(qr);      qb[0].q[1] = *(const v4u*)(qr + 16);
    qb[1].q[0] = *(const v4u*)(qr + 32); qb[1].q[1] = *(const v4u*)(qr + 48);
  }

  float m_i = -3.0e38f, l_i = 0.0f;
  v8f o[4];
  #pragma unroll
  for (int i = 0; i < 4; ++i) o[i] = {};

  #pragma unroll 1
  for (int kb = 0; kb < NK; kb += 32) {
    // ---- S^T tiles (keys = M, queries = N), 2 x (16x16), K-dim = 64 ----
    v8f s[2];
    #pragma unroll
    for (int t = 0; t < 2; ++t) {
      const unsigned short* kr = Kbase + (size_t)(kb + 16 * t + l15) * HDIM + akb;
      Frag16 ka0, ka1;
      ka0.q[0] = *(const v4u*)(kr);       ka0.q[1] = *(const v4u*)(kr + 16);
      ka1.q[0] = *(const v4u*)(kr + 32);  ka1.q[1] = *(const v4u*)(kr + 48);
      v8f st = {};
      st = __builtin_amdgcn_wmma_f32_16x16x32_bf16(false, ka0.v, false, qb[0].v, (short)0, st, false, false);
      st = __builtin_amdgcn_wmma_f32_16x16x32_bf16(false, ka1.v, false, qb[1].v, (short)0, st, false, false);
      s[t] = st;
    }
    // ---- scale + key-validity mask (key = kb + 16t + r + 8*half) ----
    #pragma unroll
    for (int t = 0; t < 2; ++t) {
      const int* mp = mrow + kb + 16 * t + 8 * half;
      v4i mv0 = *(const v4i*)(mp);
      v4i mv1 = *(const v4i*)(mp + 4);
      #pragma unroll
      for (int r = 0; r < 8; ++r) {
        int mv = (r < 4) ? mv0[r] : mv1[r - 4];
        s[t][r] = mv ? (s[t][r] * SCALE) : NEG_INF;
      }
    }
    // ---- online softmax (per-query stats are per-lane scalars) ----
    float cm = s[0][0];
    #pragma unroll
    for (int r = 1; r < 8; ++r) cm = fmaxf(cm, s[0][r]);
    #pragma unroll
    for (int r = 0; r < 8; ++r) cm = fmaxf(cm, s[1][r]);
    cm = fmaxf(cm, __shfl_xor(cm, 16, 32));
    float m_new = fmaxf(m_i, cm);
    float alpha = __expf(m_i - m_new);

    float p0[8], p1[8], ls = 0.0f;
    #pragma unroll
    for (int r = 0; r < 8; ++r) { p0[r] = __expf(s[0][r] - m_new); ls += p0[r]; }
    #pragma unroll
    for (int r = 0; r < 8; ++r) { p1[r] = __expf(s[1][r] - m_new); ls += p1[r]; }
    ls += __shfl_xor(ls, 16, 32);
    l_i = l_i * alpha + ls;
    m_i = m_new;

    #pragma unroll
    for (int i = 0; i < 4; ++i)
      #pragma unroll
      for (int r = 0; r < 8; ++r) o[i][r] *= alpha;

    // ---- build P^T B-fragment (32 keys x 16 queries) via xor-16 shuffles ----
    float p0x[8], p1x[8];
    #pragma unroll
    for (int r = 0; r < 8; ++r) {
      p0x[r] = __shfl_xor(p0[r], 16, 32);
      p1x[r] = __shfl_xor(p1[r], 16, 32);
    }
    Frag16 pb;
    #pragma unroll
    for (int j = 0; j < 8; ++j) {
      pb.v[j]     = (__bf16)(half ? p1x[j] : p0[j]);
      pb.v[8 + j] = (__bf16)(half ? p1[j] : p0x[j]);
    }

    // ---- O^T += V^T * P^T  (4 head-dim tiles of 16) ----
    #pragma unroll
    for (int i = 0; i < 4; ++i) {
      const unsigned short* vr = Vbase + (size_t)(16 * i + l15) * NK + kb + (half ? 16 : 0);
      Frag16 va;
      va.q[0] = *(const v4u*)(vr);
      va.q[1] = *(const v4u*)(vr + 8);
      o[i] = __builtin_amdgcn_wmma_f32_16x16x32_bf16(false, va.v, false, pb.v, (short)0, o[i], false, false);
    }
  }

  const float inv = 1.0f / l_i;
  #pragma unroll
  for (int i = 0; i < 4; ++i)
    #pragma unroll
    for (int r = 0; r < 8; ++r) {
      int dh = 16 * i + r + 8 * half;
      int qq = q0 + l15;
      O[(size_t)(b * NQ + qq) * DIM + h * HDIM + dh] = bfbits(o[i][r] * inv);
    }
}

// ---------------------------------------------------------------------------
extern "C" void kernel_launch(void* const* d_in, const int* in_sizes, int n_in,
                              void* d_out, int out_size, void* d_ws, size_t ws_size,
                              hipStream_t stream) {
  const float* q    = (const float*)d_in[0];
  const float* k    = (const float*)d_in[1];
  const float* v    = (const float*)d_in[2];
  const int*   mask = (const int*)d_in[3];
  const float* Wq   = (const float*)d_in[4];
  const float* Wk   = (const float*)d_in[5];
  const float* Wv   = (const float*)d_in[6];
  const float* Wp   = (const float*)d_in[7];
  const float* bp   = (const float*)d_in[8];

  const size_t actN = (size_t)BATCH * NQ * DIM;   // 8,388,608 (activations)
  const size_t wN   = (size_t)DIM * DIM;          // 1,048,576 (weights)

  unsigned short* Qh  = (unsigned short*)d_ws;    // [B,H,NQ,64]
  unsigned short* Kh  = Qh  + actN;               // [B,H,NK,64]
  unsigned short* Vt  = Kh  + actN;               // [B,H,64,NK]
  unsigned short* qb  = Vt  + actN;               // bf16 activations
  unsigned short* kb  = qb  + actN;
  unsigned short* vb  = kb  + actN;
  unsigned short* Wqb = vb  + actN;               // bf16 weights
  unsigned short* Wkb = Wqb + wN;
  unsigned short* Wvb = Wkb + wN;
  unsigned short* Wpb = Wvb + wN;
  unsigned short* Ao  = qb;                       // alias: qb dead after proj Q

  dim3 blk(256);

  // one-pass fp32 -> bf16 conversions
  cvt_bf16_kernel<<<dim3((int)(actN / 2048)), blk, 0, stream>>>(q,  qb,  (int)(actN / 8));
  cvt_bf16_kernel<<<dim3((int)(actN / 2048)), blk, 0, stream>>>(k,  kb,  (int)(actN / 8));
  cvt_bf16_kernel<<<dim3((int)(actN / 2048)), blk, 0, stream>>>(v,  vb,  (int)(actN / 8));
  cvt_bf16_kernel<<<dim3((int)(wN   / 2048)), blk, 0, stream>>>(Wq, Wqb, (int)(wN / 8));
  cvt_bf16_kernel<<<dim3((int)(wN   / 2048)), blk, 0, stream>>>(Wk, Wkb, (int)(wN / 8));
  cvt_bf16_kernel<<<dim3((int)(wN   / 2048)), blk, 0, stream>>>(Wv, Wvb, (int)(wN / 8));
  cvt_bf16_kernel<<<dim3((int)(wN   / 2048)), blk, 0, stream>>>(Wp, Wpb, (int)(wN / 8));

  // projections: 4096 waves of 32x64 tiles -> 512 blocks
  dim3 gemm_grid(512);
  gemm_bf16_kernel<<<gemm_grid, blk, 0, stream>>>(qb, Wqb, Qh, nullptr, nullptr, 0);
  gemm_bf16_kernel<<<gemm_grid, blk, 0, stream>>>(kb, Wkb, Kh, nullptr, nullptr, 0);
  gemm_bf16_kernel<<<gemm_grid, blk, 0, stream>>>(vb, Wvb, Vt, nullptr, nullptr, 1);

  // attention: 8192 waves (64 heads x 128 query tiles) -> 1024 blocks
  attn_kernel<<<dim3(1024), blk, 0, stream>>>(Qh, Kh, Vt, mask, Ao);

  // output projection -> fp32 d_out
  gemm_bf16_kernel<<<gemm_grid, blk, 0, stream>>>(Ao, Wpb, nullptr, (float*)d_out, bp, 2);
}
